// INNModel_26826365730968
// MI455X (gfx1250) — matrix operand order repaired and checked by reading, not compiled, not run on hardware
//
#include <hip/hip_runtime.h>
#include <cstdint>
#include <cstddef>

// ---------------- problem constants (from reference) ----------------
#define SEQ   512
#define BATCH 32
#define EDIM  256
#define HID   256
#define H2    512          // 2*HID
#define G4    1024         // 4*HID gates
#define NE    64
#define NS    64
#define NNODE 128          // NE + NS
#define SW    8
#define RDIM  128
#define MROWS (SEQ*BATCH)  // 16384

// ---------------- CDNA5 WMMA types ----------------
typedef __attribute__((ext_vector_type(16))) __bf16 v16bf;
typedef __attribute__((ext_vector_type(8)))  float  v8f;

// fp32 -> bf16 (round to nearest even), stored as raw u16 so we never do __bf16 scalar math
__device__ __forceinline__ uint16_t f2bf(float f) {
  uint32_t u = __float_as_uint(f);
  if ((u & 0x7F800000u) != 0x7F800000u)          // not inf/nan
    u = u + 0x7FFFu + ((u >> 16) & 1u);
  return (uint16_t)(u >> 16);
}

__device__ __forceinline__ float sigmoidf(float x) {
  return 1.0f / (1.0f + __expf(-x));
}

union BfVec { v16bf v; uint32_t u[8]; };

// A operand (16x32 bf16, MxK): lane<16 -> row=lane, k in [k0, k0+8)u[0..3], [k0+16,k0+24)u[4..7]
//                              lane>=16 -> row=lane-16, k offset +8
__device__ __forceinline__ v16bf load_a16(const uint16_t* mat, int ld, int k0, int lane) {
  const uint16_t* p = mat + (size_t)(lane & 15) * ld + (k0 + ((lane & 16) ? 8 : 0));
  const uint32_t* q0 = (const uint32_t*)p;
  const uint32_t* q1 = (const uint32_t*)(p + 16);
  BfVec t;
  t.u[0] = q0[0]; t.u[1] = q0[1]; t.u[2] = q0[2]; t.u[3] = q0[3];
  t.u[4] = q1[0]; t.u[5] = q1[1]; t.u[6] = q1[2]; t.u[7] = q1[3];
  return t.v;
}

// B operand (32x16 bf16, KxN) fed from W stored row-major [N,K] (i.e. B = W^T):
// lane<16 -> col=lane, k in [k0,k0+8) u[0..3], [k0+16,k0+24) u[4..7]; lane>=16 -> col=lane-16, k offset +8.
__device__ __forceinline__ v16bf load_b16(const uint16_t* wrow0, int ld, int k0, int lane) {
  const uint16_t* p = wrow0 + (size_t)(lane & 15) * ld + (k0 + ((lane & 16) ? 8 : 0));
  const uint32_t* q = (const uint32_t*)p;
  BfVec t;
  t.u[0] = q[0]; t.u[1] = q[1]; t.u[2] = q[2]; t.u[3] = q[3];
  t.u[4] = q[8]; t.u[5] = q[9]; t.u[6] = q[10]; t.u[7] = q[11];
  return t.v;
}

__device__ __forceinline__ v8f wmma_bf16(v16bf a, v16bf b, v8f c) {
  return __builtin_amdgcn_wmma_f32_16x16x32_bf16(false, a, false, b, (short)0, c, false, false);
}

// ---------------- K0: generic f32 -> bf16 convert ----------------
__global__ void k_f2bf(const float* __restrict__ src, uint16_t* __restrict__ dst, int n) {
  int i = blockIdx.x * blockDim.x + threadIdx.x;
  if (i < n) dst[i] = f2bf(src[i]);
}

// ---------------- K1: embedding gather + bf16 convert ----------------
// emb_bf row-major [SEQ*BATCH, EDIM], row = s*BATCH + b
__global__ void k_embed(const int* __restrict__ tokens, const float* __restrict__ word_emb,
                        uint16_t* __restrict__ emb_bf) {
  int row = blockIdx.x;                 // 0..16383
  int tok = tokens[row];
  const float* src = word_emb + (size_t)tok * EDIM;
  uint16_t* dst = emb_bf + (size_t)row * EDIM;
  for (int e = threadIdx.x; e < EDIM; e += blockDim.x) dst[e] = f2bf(src[e]);
}

// ---------------- K2: X-gate GEMM  X = emb @ Wih^T + b  (both directions) ----------------
// grid (Mtiles=1024, 2, dir=2), block 256 = 8 waves; each wave: one 16-row tile x 4 N-tiles.
__global__ __launch_bounds__(256)
void k_xgemm(const uint16_t* __restrict__ emb_bf,
             const uint16_t* __restrict__ wih_f, const uint16_t* __restrict__ wih_b,
             const float* __restrict__ bl_f, const float* __restrict__ bl_b,
             float* __restrict__ Xf, float* __restrict__ Xb) {
  const int dir  = blockIdx.z;
  const uint16_t* W = dir ? wih_b : wih_f;
  const float* bl   = dir ? bl_b  : bl_f;
  float* X          = dir ? Xb    : Xf;

  const int lane   = threadIdx.x & 31;
  const int wave   = threadIdx.x >> 5;
  const int ngroup = blockIdx.y * 8 + wave;          // 0..15 -> 4 N-tiles each
  const int mtile  = blockIdx.x;                     // 0..1023
  const uint16_t* Abase = emb_bf + (size_t)(mtile * 16) * EDIM;

  v8f acc[4];
  for (int k = 0; k < 4; ++k) {
    int n = (ngroup * 4 + k) * 16 + (lane & 15);
    float bv = bl[n];
    for (int v = 0; v < 8; ++v) acc[k][v] = bv;      // bias rides in the accumulator
  }

  for (int k0 = 0; k0 < EDIM; k0 += 32) {
    v16bf a = load_a16(Abase, EDIM, k0, lane);
    for (int k = 0; k < 4; ++k) {
      const uint16_t* Bbase = W + (size_t)((ngroup * 4 + k) * 16) * EDIM;
      if (k0 + 32 < EDIM)
        __builtin_prefetch(Bbase + (size_t)(lane & 15) * EDIM + k0 + 32, 0, 1);
      v16bf b = load_b16(Bbase, EDIM, k0, lane);
      acc[k] = wmma_bf16(a, b, acc[k]);
    }
  }

  // C/D layout: VGPR v, lane l -> M = v + (l<16?0:8), N = l%16
  const int mrow = mtile * 16 + ((lane & 16) ? 8 : 0);
  for (int k = 0; k < 4; ++k) {
    int n = (ngroup * 4 + k) * 16 + (lane & 15);
    for (int v = 0; v < 8; ++v)
      X[(size_t)(mrow + v) * G4 + n] = acc[k][v];
  }
}

// ---------------- K3: recurrent BiLSTM, one workgroup per direction ----------------
// 1024 threads = 32 waves fill a WGP. h (bf16) and c (f32) persist in LDS.
// Wave w: mt = w>>4 (batches 16mt..16mt+15), col = w&15 -> hidden cols [16col,16col+16)
// and it owns the SAME column tile in all 4 gates -> activations stay in registers.
__global__ __launch_bounds__(1024)
void k_lstm(const float* __restrict__ Xf, const float* __restrict__ Xb,
            const uint16_t* __restrict__ whh_f, const uint16_t* __restrict__ whh_b,
            float* __restrict__ blstm) {
  const int dir = blockIdx.x;
  const float* X    = dir ? Xb    : Xf;
  const uint16_t* W = dir ? whh_b : whh_f;

  __shared__ __align__(16) uint16_t hsh[BATCH * HID];   // 16 KB
  __shared__ __align__(16) float    csh[BATCH * HID];   // 32 KB

  const int tid  = threadIdx.x;
  const int lane = tid & 31;
  const int wave = tid >> 5;
  const int mt   = wave >> 4;          // 0..1
  const int col  = wave & 15;          // 0..15

  for (int i = tid; i < BATCH * HID; i += 1024) { hsh[i] = 0; csh[i] = 0.0f; }
  __syncthreads();

  for (int step = 0; step < SEQ; ++step) {
    const int t = dir ? (SEQ - 1 - step) : step;
    const float* xrow = X + (size_t)(t * BATCH) * G4;

    // ---- GEMM phase: G = X[t] + h @ Whh^T, 4 gate tiles per wave ----
    v8f acc[4];
    const int mrow = mt * 16 + ((lane & 16) ? 8 : 0);
    for (int g = 0; g < 4; ++g) {
      int n = g * HID + col * 16 + (lane & 15);
      for (int v = 0; v < 8; ++v) acc[g][v] = xrow[(size_t)(mrow + v) * G4 + n];
    }
    const uint16_t* Ah = hsh + (size_t)(mt * 16) * HID;   // LDS-resident h
    for (int k0 = 0; k0 < HID; k0 += 32) {
      v16bf a = load_a16(Ah, HID, k0, lane);
      for (int g = 0; g < 4; ++g) {
        const uint16_t* Bbase = W + (size_t)(g * HID + col * 16) * HID;
        v16bf b = load_b16(Bbase, HID, k0, lane);
        acc[g] = wmma_bf16(a, b, acc[g]);
      }
    }
    __syncthreads();   // all waves done reading h before anyone rewrites it

    // ---- pointwise LSTM cell entirely from this wave's accumulators ----
    const int j = col * 16 + (lane & 15);                // hidden index 0..255
    for (int v = 0; v < 8; ++v) {
      int m  = mt * 16 + v + ((lane & 16) ? 8 : 0);      // batch row 0..31
      float ig = sigmoidf(acc[0][v]);
      float fg = sigmoidf(acc[1][v]);
      float gg = tanhf(acc[2][v]);
      float og = sigmoidf(acc[3][v]);
      float c  = fg * csh[m * HID + j] + ig * gg;
      csh[m * HID + j] = c;
      float h = og * tanhf(c);
      hsh[m * HID + j] = f2bf(h);
      blstm[((size_t)t * BATCH + m) * H2 + dir * HID + j] = h;
    }
    __syncthreads();   // h/c updated before next step's WMMA reads
  }
}

// ---------------- K4: attention scores attn[s,b] = blstm[s,b,:].w + b ----------------
__global__ void k_attn(const float* __restrict__ blstm, const float* __restrict__ attn_w,
                       const float* __restrict__ attn_b, float* __restrict__ attn) {
  int idx = blockIdx.x * blockDim.x + threadIdx.x;     // s*BATCH + b
  if (idx >= SEQ * BATCH) return;
  const float* r = blstm + (size_t)idx * H2;
  float acc = attn_b[0];
  for (int k = 0; k < H2; ++k) acc += r[k] * attn_w[k];
  attn[idx] = acc;
}

// ---------------- K5: span softmax-sum + H init ----------------
__global__ void k_hinit(const float* __restrict__ attn, const int* __restrict__ spans,
                        const float* __restrict__ blstm, float* __restrict__ Hws) {
  const int node = blockIdx.x;   // 0..127
  const int b    = blockIdx.y;   // 0..31
  float* dst = Hws + ((size_t)node * BATCH + b) * H2;
  if (node < NE) {
    __shared__ float wsum;
    if (threadIdx.x == 0) {
      float s[SW], m = -3.4e38f;
      for (int i = 0; i < SW; ++i) { s[i] = attn[spans[node * SW + i] * BATCH + b]; m = fmaxf(m, s[i]); }
      float tot = 0.0f;
      for (int i = 0; i < SW; ++i) tot += __expf(s[i] - m);
      float ws = 0.0f;
      for (int i = 0; i < SW; ++i) ws += __expf(s[i] - m) / tot;   // == 1 up to fp, as in reference
      wsum = ws;
    }
    __syncthreads();
    const float* src = blstm + ((size_t)node * BATCH + b) * H2;    // seq position == node
    for (int j = threadIdx.x; j < H2; j += blockDim.x) dst[j] = wsum * src[j];
  } else {
    for (int j = threadIdx.x; j < H2; j += blockDim.x) dst[j] = 0.0f;
  }
}

// ---------------- K6: sequential DAG-LSTM (2048-step chain), one persistent workgroup ----------------
__global__ __launch_bounds__(1024)
void k_dag(const int* __restrict__ S, const int* __restrict__ T, const int* __restrict__ elnames,
           const int* __restrict__ entsz, const float* __restrict__ elem_emb,
           const float* __restrict__ W_ioc, const float* __restrict__ U_ioc, const float* __restrict__ b_ioc,
           const float* __restrict__ Wfm, const float* __restrict__ Ufm, const float* __restrict__ bfv,
           const float* __restrict__ W1, const float* __restrict__ b1,
           const float* __restrict__ W2, const float* __restrict__ b2,
           float* __restrict__ Hws, float* __restrict__ Pout) {
  __shared__ float hv[2 * H2];        // concat(H[a0], H[a1])
  __shared__ float evec[RDIM];
  __shared__ float fg[2 * H2];        // per-input forget gates
  __shared__ float ioc[3 * H2];
  __shared__ float cst[H2];           // rolling cell state (across batches!)
  __shared__ float cnew[H2];
  __shared__ float hout[H2];
  __shared__ float hid1[2 * H2];
  __shared__ float Pb[NNODE][2];
  __shared__ float logit[2];
  __shared__ int   sc[6];

  const int tid = threadIdx.x;
  if (tid < H2) cst[tid] = 0.0f;
  __syncthreads();

  for (int b = 0; b < BATCH; ++b) {
    if (tid < NNODE) { Pb[tid][0] = (tid < NE) ? 0.001f : 0.0f;
                       Pb[tid][1] = (tid < NE) ? 0.999f : 0.0f; }
    __syncthreads();

    for (int n = 0; n < NS; ++n) {
      if (tid == 0) {
        int a0 = S[(n * BATCH + b) * 2 + 0];
        int a1 = S[(n * BATCH + b) * 2 + 1];
        int tg = T[n * BATCH + b];
        int el = elnames[n * BATCH + b];
        int valid = (tg >= entsz[b]) && (el > -1);
        int cond  = valid && (Pb[a0][1] > 0.5f) && (Pb[a1][1] > 0.5f);
        sc[0] = a0; sc[1] = a1; sc[2] = tg; sc[3] = cond; sc[4] = valid;
        sc[5] = (el < 0) ? 0 : el;
      }
      __syncthreads();
      const int a0 = sc[0], a1 = sc[1], tg = sc[2], cond = sc[3], valid = sc[4], el = sc[5];

      // gather hv = [H[a0,b,:], H[a1,b,:]], e = elem_emb[el]
      if (tid < H2)      hv[tid] = Hws[((size_t)a0 * BATCH + b) * H2 + tid];
      else               hv[tid] = Hws[((size_t)a1 * BATCH + b) * H2 + (tid - H2)];
      if (tid < RDIM)    evec[tid] = elem_emb[(size_t)el * RDIM + tid];
      __syncthreads();

      // forget gates: f[half][j] = sigmoid(Wf[j].e + Uf[j].hv[half] + bf[j])   (1024 rows)
      {
        const int half = tid >> 9, r = tid & (H2 - 1);
        float acc = bfv[r];
        const float* wr = Wfm + (size_t)r * RDIM;
        for (int k = 0; k < RDIM; ++k) acc += wr[k] * evec[k];
        const float* ur = Ufm + (size_t)r * H2;
        const float* hh = hv + half * H2;
        for (int k = 0; k < H2; ++k) acc += ur[k] * hh[k];
        fg[tid] = sigmoidf(acc);
      }
      // i/o/c-hat rows (1536): thread r does rows r and r+1024 (r<512)
      for (int rr = tid; rr < 3 * H2; rr += 1024) {
        float acc = b_ioc[rr];
        const float* wr = W_ioc + (size_t)rr * RDIM;
        for (int k = 0; k < RDIM; ++k) acc += wr[k] * evec[k];
        const float* ur = U_ioc + (size_t)rr * (2 * H2);
        for (int k = 0; k < 2 * H2; ++k) acc += ur[k] * hv[k];
        ioc[rr] = acc;
      }
      __syncthreads();

      if (tid < H2) {
        float ig = sigmoidf(ioc[tid]);
        float ch = tanhf(ioc[2 * H2 + tid]);
        float c2 = (fg[tid] + fg[H2 + tid]) * cst[tid] + ig * ch;
        c2 = fminf(fmaxf(c2, -100.0f), 100.0f);
        cnew[tid] = c2;
        float og = sigmoidf(ioc[H2 + tid]);
        float h = og * tanhf(c2);
        hout[tid] = fminf(fmaxf(h, -100.0f), 100.0f);
      }
      __syncthreads();

      // hid1 = W1 @ h_out + b1  (1024 rows)
      {
        float acc = b1[tid];
        const float* wr = W1 + (size_t)tid * H2;
        for (int k = 0; k < H2; ++k) acc += wr[k] * hout[k];
        hid1[tid] = acc;
      }
      __syncthreads();
      if (tid < 2) {
        float acc = b2[tid];
        const float* wr = W2 + (size_t)tid * (2 * H2);
        for (int k = 0; k < 2 * H2; ++k) acc += wr[k] * hid1[k];
        logit[tid] = acc;
      }
      __syncthreads();
      if (tid == 0) {
        float m = fmaxf(logit[0], logit[1]);
        float e0 = __expf(logit[0] - m), e1 = __expf(logit[1] - m);
        float s = e0 + e1;
        float p0 = cond ? (e0 / s) : 0.999f;   // rej = [0.999, 0.001]
        float p1 = cond ? (e1 / s) : 0.001f;
        if (valid) { Pb[tg][0] = p0; Pb[tg][1] = p1; }
      }
      if (tid < H2 && cond) {
        Hws[((size_t)tg * BATCH + b) * H2 + tid] = hout[tid];
        cst[tid] = cnew[tid];                  // c rolls only when cond
      }
      __syncthreads();
    }

    if (tid < NNODE * 2) Pout[b * (NNODE * 2) + tid] = Pb[tid >> 1][tid & 1];
    __syncthreads();
  }
}

// ---------------- host side ----------------
extern "C" void kernel_launch(void* const* d_in, const int* in_sizes, int n_in,
                              void* d_out, int out_size, void* d_ws, size_t ws_size,
                              hipStream_t stream) {
  const int*   tokens   = (const int*)  d_in[0];
  const int*   spans    = (const int*)  d_in[1];
  const int*   elnames  = (const int*)  d_in[2];
  /* d_in[3] = H (zeros, rebuilt in ws), d_in[4] = A (unused) */
  const int*   T        = (const int*)  d_in[5];
  const int*   S        = (const int*)  d_in[6];
  const int*   entsz    = (const int*)  d_in[7];
  const float* word_emb = (const float*)d_in[8];
  const float* elem_emb = (const float*)d_in[9];
  const float* attn_w   = (const float*)d_in[10];
  const float* attn_b   = (const float*)d_in[11];
  const float* Wih_f    = (const float*)d_in[12];
  const float* Whh_f    = (const float*)d_in[13];
  const float* bl_f     = (const float*)d_in[14];
  const float* Wih_b    = (const float*)d_in[15];
  const float* Whh_b    = (const float*)d_in[16];
  const float* bl_b     = (const float*)d_in[17];
  const float* W_ioc    = (const float*)d_in[18];
  const float* U_ioc    = (const float*)d_in[19];
  const float* b_ioc    = (const float*)d_in[20];
  const float* Wfm      = (const float*)d_in[21];
  const float* Ufm      = (const float*)d_in[22];
  const float* bfv      = (const float*)d_in[23];
  const float* W1       = (const float*)d_in[24];
  const float* b1       = (const float*)d_in[25];
  const float* W2       = (const float*)d_in[26];
  const float* b2       = (const float*)d_in[27];
  (void)in_sizes; (void)n_in; (void)out_size; (void)ws_size;

  // workspace carve-up
  char* w = (char*)d_ws;
  size_t off = 0;
  auto carve = [&](size_t bytes) { void* p = w + off; off = (off + bytes + 255) & ~(size_t)255; return p; };
  uint16_t* emb_bf  = (uint16_t*)carve((size_t)MROWS * EDIM * 2);       // 8 MB
  uint16_t* wihf_bf = (uint16_t*)carve((size_t)G4 * EDIM * 2);          // 512 KB x4
  uint16_t* whhf_bf = (uint16_t*)carve((size_t)G4 * HID * 2);
  uint16_t* wihb_bf = (uint16_t*)carve((size_t)G4 * EDIM * 2);
  uint16_t* whhb_bf = (uint16_t*)carve((size_t)G4 * HID * 2);
  float*    Xf      = (float*)   carve((size_t)MROWS * G4 * 4);         // 64 MB
  float*    Xb      = (float*)   carve((size_t)MROWS * G4 * 4);         // 64 MB
  float*    blstm   = (float*)   carve((size_t)MROWS * H2 * 4);         // 32 MB
  float*    attn    = (float*)   carve((size_t)MROWS * 4);              // 64 KB
  float*    Hws     = (float*)   carve((size_t)NNODE * BATCH * H2 * 4); // 8 MB

  const int NW = G4 * EDIM;   // 262144 elements per LSTM weight matrix
  hipLaunchKernelGGL(k_f2bf, dim3((NW + 255) / 256), dim3(256), 0, stream, Wih_f, wihf_bf, NW);
  hipLaunchKernelGGL(k_f2bf, dim3((NW + 255) / 256), dim3(256), 0, stream, Whh_f, whhf_bf, NW);
  hipLaunchKernelGGL(k_f2bf, dim3((NW + 255) / 256), dim3(256), 0, stream, Wih_b, wihb_bf, NW);
  hipLaunchKernelGGL(k_f2bf, dim3((NW + 255) / 256), dim3(256), 0, stream, Whh_b, whhb_bf, NW);

  hipLaunchKernelGGL(k_embed, dim3(MROWS), dim3(256), 0, stream, tokens, word_emb, emb_bf);

  hipLaunchKernelGGL(k_xgemm, dim3(MROWS / 16, 2, 2), dim3(256), 0, stream,
                     emb_bf, wihf_bf, wihb_bf, bl_f, bl_b, Xf, Xb);

  hipLaunchKernelGGL(k_lstm, dim3(2), dim3(1024), 0, stream, Xf, Xb, whhf_bf, whhb_bf, blstm);

  hipLaunchKernelGGL(k_attn, dim3(MROWS / 256), dim3(256), 0, stream, blstm, attn_w, attn_b, attn);

  hipLaunchKernelGGL(k_hinit, dim3(NNODE, BATCH), dim3(256), 0, stream, attn, spans, blstm, Hws);

  hipLaunchKernelGGL(k_dag, dim3(1), dim3(1024), 0, stream,
                     S, T, elnames, entsz, elem_emb,
                     W_ioc, U_ioc, b_ioc, Wfm, Ufm, bfv, W1, b1, W2, b2,
                     Hws, (float*)d_out);
}
